// MultiheadAttention_51049981280681
// MI455X (gfx1250) — matrix operand order repaired and checked by reading, not compiled, not run on hardware
//
#include <hip/hip_runtime.h>
#include <hip/hip_bf16.h>

// ---------------------------------------------------------------------------
// Causal multi-head attention forward for MI455X (gfx1250, wave32, WMMA+TDM).
//   x:[2,2048,1024] f32, WQ/WK/WV/WO:[1024,1024] f32 -> out:[2,2048,1024] f32
// Matmuls use v_wmma_f32_16x16x32_bf16 (f32 accumulate). Attention is
// flash-style (online softmax). Attention Q/K tiles are fetched with the
// Tensor Data Mover (tensor_load_to_lds + s_wait_tensorcnt) when the builtin
// is available, using the D# pad feature to produce the padded LDS layout.
// ---------------------------------------------------------------------------

typedef __attribute__((ext_vector_type(16))) __bf16       v16bf;
typedef __attribute__((ext_vector_type(8)))  float        v8f;
typedef __attribute__((ext_vector_type(4)))  unsigned int u32x4;
typedef __attribute__((ext_vector_type(8)))  int          i32x8;
typedef __attribute__((ext_vector_type(4)))  int          i32x4;

#define D_MODEL   1024
#define NUM_HEADS 16
#define D_K       64
#define BATCH     2
#define SEQ       2048
#define MTOT      (BATCH * SEQ)   // 4096 rows

#if defined(__has_builtin)
#if __has_builtin(__builtin_amdgcn_tensor_load_to_lds) && \
    __has_builtin(__builtin_amdgcn_s_wait_tensorcnt)
#define USE_TDM 1
#endif
#endif

// ---- bf16 helpers ---------------------------------------------------------
__device__ __forceinline__ __bf16 f2bf(float f) {
  unsigned u = __builtin_bit_cast(unsigned, f);
  unsigned r = (u + 0x7FFFu + ((u >> 16) & 1u)) >> 16;   // round-nearest-even
  unsigned short s = (unsigned short)r;
  return __builtin_bit_cast(__bf16, s);
}

__device__ __forceinline__ v8f wmma_bf16(v16bf a, v16bf b, v8f c) {
  return __builtin_amdgcn_wmma_f32_16x16x32_bf16(
      /*neg_a=*/false, a, /*neg_b=*/false, b,
      /*c_mod=*/(short)0, c, /*reuse_a=*/false, /*reuse_b=*/false);
}

// A fragment: 16x32 bf16 from row-major LDS tile. lane: m=lane&15, kh=lane>>4.
// elem i -> K = k0 + kh*16 + i.
__device__ __forceinline__ v16bf load_fragA(const __bf16* base, int ld,
                                            int row, int kh, int k0) {
  const __bf16* p0 = base + row * ld + k0 + kh * 16;
  v16bf r;
#pragma unroll
  for (int i = 0; i < 16; ++i) r[i] = p0[i];
  return r;
}

// B fragment: 32x16 bf16 from an N-major ("B-transposed") LDS tile Bt[n][k].
// lane: n = lane&15, kh = lane>>4.  elem i -> K = k0 + kh*16 + i.
__device__ __forceinline__ v16bf load_fragB(const __bf16* bt, int ld,
                                            int col, int kh, int k0) {
  const __bf16* p = bt + col * ld + k0 + kh * 16;
  v16bf r;
#pragma unroll
  for (int i = 0; i < 16; ++i) r[i] = p[i];
  return r;
}

#ifdef USE_TDM
// 2D TDM tile load (bf16 elements) with LDS row padding:
// tile rows of 64 elems (32 DWORDs) padded by 4 DWORDs -> 72-elem LDS rows.
// D# bit packing per CDNA5 ISA.
// This toolchain declares the 6-arg builtin:
//   (uint32x4 g0, int32x8 g1, int32x4 g2, int32x4 g3, int32x8, i32 cpol)
__device__ __forceinline__ void tdm_load_2d_64x64(unsigned lds_byte_off,
                                                  const __bf16* gptr) {
  unsigned long long ga = (unsigned long long)(const void*)gptr;
  u32x4 g0;
  g0[0] = 1u;                                             // count=1 (user D#)
  g0[1] = lds_byte_off;                                   // lds_addr
  g0[2] = (unsigned)(ga & 0xFFFFFFFFu);                   // global_addr lo
  g0[3] = (unsigned)((ga >> 32) & 0x01FFFFFFu) | (2u << 30); // addr hi | type=2
  i32x8 g1;
  // mask=0 | data_size=1(2B) | pad_enable | pad_interval=4(32dw) | pad_amount=3(4dw)
  g1[0] = (int)((1u << 16) | (1u << 20) | (4u << 22) | (3u << 25));
  g1[1] = (int)((unsigned)(D_K & 0xFFFF) << 16);          // tensor_dim0 lo16
  g1[2] = (int)(((unsigned)D_K >> 16) | (64u << 16));     // dim0 hi | tensor_dim1 lo
  g1[3] = (int)((64u >> 16) | ((unsigned)D_K << 16));     // dim1 hi | tile_dim0
  g1[4] = (int)64u;                                       // tile_dim1 | tile_dim2=0
  g1[5] = (int)(unsigned)D_K;                             // tensor_dim0_stride lo
  g1[6] = 0;                                              // stride0 hi | stride1 lo
  g1[7] = 0;                                              // stride1 hi
  i32x4 gz4 = {0, 0, 0, 0};                               // 2D: groups 2/3 unused
  i32x8 gz8 = {0, 0, 0, 0, 0, 0, 0, 0};
  __builtin_amdgcn_tensor_load_to_lds(g0, g1, gz4, gz4, gz8, 0);
}
#endif

// ---- fp32 -> bf16 convert -------------------------------------------------
__global__ __launch_bounds__(256) void cvt_f32_bf16(const float* __restrict__ src,
                                                    __bf16* __restrict__ dst, int n) {
  int i = blockIdx.x * blockDim.x + threadIdx.x;
  if (i < n) dst[i] = f2bf(src[i]);
}

// ---- WMMA GEMM: 128x128 tile, 256 thr (8 waves, 2x4), K-step 32 -----------
// Per wave: 64x32 output = 4x2 WMMA tiles -> 8 wmma per 6 fragment loads.
// MODE 0: bf16 out remapped to [B,H,S,64] (QKV). MODE 1: f32 out row-major.
#define GT_M 128
#define GT_N 128
#define GT_K 32
#define LDT  (GT_K + 8)   // 40 elems = 80B rows (16B multiple)

template <int MODE>
__global__ __launch_bounds__(256) void gemm_bf16_128x128(
    const __bf16* __restrict__ A, const __bf16* __restrict__ B,
    void* __restrict__ Cout, int M, int N, int K) {
  __shared__ __bf16 As[GT_M][LDT];
  __shared__ __bf16 Bts[GT_N][LDT];   // N-major: Bts[n][k]

  const int tid   = threadIdx.x;
  const int lane  = tid & 31;
  const int wave  = tid >> 5;
  const int waveM = wave & 1;          // 2 waves along M -> 64 rows each
  const int waveN = wave >> 1;         // 4 waves along N -> 32 cols each
  const int n16   = lane & 15;
  const int kh    = lane >> 4;
  const int m0    = blockIdx.x * GT_M;
  const int n0    = blockIdx.y * GT_N;

  const int arow  = tid >> 1;          // 0..127
  const int aseg  = (tid & 1) * 16;    // 0 / 16 elems
  const int brow0 = tid >> 4;          // 0..15 (plus +16 row)
  const int bcol0 = (tid & 15) * 8;    // 0..120

  v8f acc[4][2] = {};

  for (int k0 = 0; k0 < K; k0 += GT_K) {
    // A tile: 128x32, two 16B loads per thread
    const uint4* ga = reinterpret_cast<const uint4*>(
        A + (size_t)(m0 + arow) * K + k0 + aseg);
    uint4 a0 = ga[0];
    uint4 a1 = ga[1];
    *reinterpret_cast<uint4*>(&As[arow][aseg])     = a0;
    *reinterpret_cast<uint4*>(&As[arow][aseg + 8]) = a1;
    // B tile: 32x128 row-major -> N-major LDS (2 rows x 8 cols per thread)
#pragma unroll
    for (int r2 = 0; r2 < 32; r2 += 16) {
      uint4 bq = *reinterpret_cast<const uint4*>(
          B + (size_t)(k0 + brow0 + r2) * N + n0 + bcol0);
      const __bf16* be = reinterpret_cast<const __bf16*>(&bq);
#pragma unroll
      for (int i = 0; i < 8; ++i) Bts[bcol0 + i][brow0 + r2] = be[i];
    }

    if (k0 + GT_K < K)  // global_prefetch_b8 of next A tile
      __builtin_prefetch(A + (size_t)(m0 + arow) * K + k0 + GT_K + aseg, 0, 0);

    __syncthreads();

    v16bf af[4], bf[2];
#pragma unroll
    for (int mi = 0; mi < 4; ++mi)
      af[mi] = load_fragA(&As[0][0], LDT, waveM * 64 + mi * 16 + n16, kh, 0);
#pragma unroll
    for (int ni = 0; ni < 2; ++ni)
      bf[ni] = load_fragB(&Bts[0][0], LDT, waveN * 32 + ni * 16 + n16, kh, 0);

#pragma unroll
    for (int mi = 0; mi < 4; ++mi)
#pragma unroll
      for (int ni = 0; ni < 2; ++ni)
        acc[mi][ni] = wmma_bf16(af[mi], bf[ni], acc[mi][ni]);

    __syncthreads();
  }

  // C/D layout: VGPR r -> M = r + 8*kh (per lane-half), N = lane&15
#pragma unroll
  for (int mi = 0; mi < 4; ++mi) {
#pragma unroll
    for (int ni = 0; ni < 2; ++ni) {
#pragma unroll
      for (int r = 0; r < 8; ++r) {
        int rowg = m0 + waveM * 64 + mi * 16 + kh * 8 + r;
        int colg = n0 + waveN * 32 + ni * 16 + n16;
        float v = acc[mi][ni][r];
        if (MODE == 0) {
          __bf16* dst = (__bf16*)Cout;
          int b = rowg >> 11, s = rowg & (SEQ - 1);
          int h = colg >> 6,  d = colg & (D_K - 1);
          dst[((size_t)(b * NUM_HEADS + h) * SEQ + s) * D_K + d] = f2bf(v);
        } else {
          float* dst = (float*)Cout;
          dst[(size_t)rowg * N + colg] = v;
        }
      }
    }
  }
}

// ---- Flash attention (causal, online softmax) -----------------------------
// grid: (SEQ/64, BATCH*NUM_HEADS), 128 threads = 4 waves.
// Wave w owns query rows [q0 + 16w, q0 + 16w + 16).
#define QBLK 64
#define KBLK 64
#define ALD  72   // 144B rows (16B multiple; matches TDM pad 32dw+4dw)

#define QS_OFF 0
#define KS_OFF (QBLK * ALD)
#define VT_OFF ((QBLK + KBLK) * ALD)
#define PS_OFF ((QBLK + KBLK + D_K) * ALD)
#define SM_ELEMS ((QBLK + KBLK + D_K + QBLK) * ALD)

__global__ __launch_bounds__(128) void attn_fwd(
    const __bf16* __restrict__ Q, const __bf16* __restrict__ Km,
    const __bf16* __restrict__ V, __bf16* __restrict__ O) {
  __shared__ __bf16 smem[SM_ELEMS];
  __bf16* Qs  = smem + QS_OFF;   // [QBLK][ALD]
  __bf16* Ks  = smem + KS_OFF;   // [KBLK][ALD]  (K rows as-is == Bt for Q.K^T)
  __bf16* Vts = smem + VT_OFF;   // [D_K][ALD]   (V transposed == Bt for P.V)
  __bf16* Ps  = smem + PS_OFF;   // [QBLK][ALD]  per-wave 16-row slabs

  const int tid  = threadIdx.x;
  const int lane = tid & 31;
  const int wave = tid >> 5;           // 0..3
  const int n16  = lane & 15;
  const int kh   = lane >> 4;
  const int bh   = blockIdx.y;         // b*16 + h
  const int q0   = blockIdx.x * QBLK;

  const size_t hbase = (size_t)bh * SEQ * D_K;

  // ---- load Q tile (64x64) ----
#ifdef USE_TDM
  if (wave == 0) {
    tdm_load_2d_64x64((unsigned)(QS_OFF * 2), Q + hbase + (size_t)q0 * D_K);
    __builtin_amdgcn_s_wait_tensorcnt(0);
  }
#else
  {
    int row  = tid >> 1;
    int cseg = (tid & 1) * 32;
    const uint4* g = reinterpret_cast<const uint4*>(
        Q + hbase + (size_t)(q0 + row) * D_K + cseg);
#pragma unroll
    for (int i = 0; i < 4; ++i)
      *reinterpret_cast<uint4*>(&Qs[row * ALD + cseg + i * 8]) = g[i];
  }
#endif
  __syncthreads();

  v16bf qf0 = load_fragA(Qs, ALD, wave * 16 + n16, kh, 0);
  v16bf qf1 = load_fragA(Qs, ALD, wave * 16 + n16, kh, 32);

  float mrow[8], lrow[8];
  v8f oacc[4] = {};
#pragma unroll
  for (int r = 0; r < 8; ++r) { mrow[r] = -__builtin_inff(); lrow[r] = 0.f; }

  const int nkb = q0 / KBLK + 1;       // causal: key blocks <= query block
  for (int kb = 0; kb < nkb; ++kb) {
    __syncthreads();                   // protect Ks/Vts from previous iter use

    // ---- load K tile (TDM when available) and V tile (transposed) ----
#ifdef USE_TDM
    if (wave == 0) {
      tdm_load_2d_64x64((unsigned)(KS_OFF * 2),
                        Km + hbase + (size_t)(kb * KBLK) * D_K);
    }
#else
    {
      int row  = tid >> 1;
      int cseg = (tid & 1) * 32;
      const uint4* gk = reinterpret_cast<const uint4*>(
          Km + hbase + (size_t)(kb * KBLK + row) * D_K + cseg);
#pragma unroll
      for (int i = 0; i < 4; ++i)
        *reinterpret_cast<uint4*>(&Ks[row * ALD + cseg + i * 8]) = gk[i];
    }
#endif
    {
      int row  = tid >> 1;
      int cseg = (tid & 1) * 32;
      const uint4* gv = reinterpret_cast<const uint4*>(
          V + hbase + (size_t)(kb * KBLK + row) * D_K + cseg);
#pragma unroll
      for (int i = 0; i < 4; ++i) {
        uint4 q = gv[i];
        const __bf16* e = reinterpret_cast<const __bf16*>(&q);
#pragma unroll
        for (int j = 0; j < 8; ++j) Vts[(cseg + i * 8 + j) * ALD + row] = e[j];
      }
    }
#ifdef USE_TDM
    if (wave == 0) __builtin_amdgcn_s_wait_tensorcnt(0);
#endif
    __syncthreads();

    // ---- S = (Q . K^T) * 1/sqrt(64) ----
    v8f sc[4];
#pragma unroll
    for (int nt = 0; nt < 4; ++nt) {
      v16bf b0 = load_fragB(Ks, ALD, nt * 16 + n16, kh, 0);
      v16bf b1 = load_fragB(Ks, ALD, nt * 16 + n16, kh, 32);
      v8f c = {};
      c = wmma_bf16(qf0, b0, c);
      c = wmma_bf16(qf1, b1, c);
      sc[nt] = c;
    }

    // ---- online softmax (row stats across 16-lane N-groups) ----
    float pv[4][8], alpha[8];
#pragma unroll
    for (int r = 0; r < 8; ++r) {
      int qrow = q0 + wave * 16 + kh * 8 + r;
      float rm = -__builtin_inff();
#pragma unroll
      for (int nt = 0; nt < 4; ++nt) {
        int kcol = kb * KBLK + nt * 16 + n16;
        float s = sc[nt][r] * 0.125f;
        if (kcol > qrow) s = -__builtin_inff();
        pv[nt][r] = s;
        rm = fmaxf(rm, s);
      }
      rm = fmaxf(rm, __shfl_xor(rm, 1));
      rm = fmaxf(rm, __shfl_xor(rm, 2));
      rm = fmaxf(rm, __shfl_xor(rm, 4));
      rm = fmaxf(rm, __shfl_xor(rm, 8));
      float mnew = fmaxf(mrow[r], rm);
      alpha[r] = __expf(mrow[r] - mnew);
      float rs = 0.f;
#pragma unroll
      for (int nt = 0; nt < 4; ++nt) {
        float p = __expf(pv[nt][r] - mnew);
        pv[nt][r] = p;
        rs += p;
      }
      rs += __shfl_xor(rs, 1);
      rs += __shfl_xor(rs, 2);
      rs += __shfl_xor(rs, 4);
      rs += __shfl_xor(rs, 8);
      lrow[r] = lrow[r] * alpha[r] + rs;
      mrow[r] = mnew;
    }

#pragma unroll
    for (int dt = 0; dt < 4; ++dt)
#pragma unroll
      for (int r = 0; r < 8; ++r)
        oacc[dt][r] *= alpha[r];

    // ---- stage P (C-layout) -> LDS -> A-layout fragments (own-wave slab) --
#pragma unroll
    for (int nt = 0; nt < 4; ++nt)
#pragma unroll
      for (int r = 0; r < 8; ++r)
        Ps[(wave * 16 + kh * 8 + r) * ALD + nt * 16 + n16] = f2bf(pv[nt][r]);

    v16bf pf0 = load_fragA(Ps, ALD, wave * 16 + n16, kh, 0);
    v16bf pf1 = load_fragA(Ps, ALD, wave * 16 + n16, kh, 32);

    // ---- O += P . V ----
#pragma unroll
    for (int dt = 0; dt < 4; ++dt) {
      v16bf b0 = load_fragB(Vts, ALD, dt * 16 + n16, kh, 0);
      v16bf b1 = load_fragB(Vts, ALD, dt * 16 + n16, kh, 32);
      oacc[dt] = wmma_bf16(pf0, b0, oacc[dt]);
      oacc[dt] = wmma_bf16(pf1, b1, oacc[dt]);
    }
  }

  // ---- normalize, write attn output [B*S, E] (bf16 for final GEMM) ----
  const int b = bh >> 4, h = bh & 15;
#pragma unroll
  for (int r = 0; r < 8; ++r) {
    float inv = 1.0f / lrow[r];
    int s = q0 + wave * 16 + kh * 8 + r;
#pragma unroll
    for (int dt = 0; dt < 4; ++dt) {
      int d = dt * 16 + n16;
      O[(size_t)(b * SEQ + s) * D_MODEL + h * D_K + d] = f2bf(oacc[dt][r] * inv);
    }
  }
}

// ---------------------------------------------------------------------------
extern "C" void kernel_launch(void* const* d_in, const int* in_sizes, int n_in,
                              void* d_out, int out_size, void* d_ws, size_t ws_size,
                              hipStream_t stream) {
  (void)in_sizes; (void)n_in; (void)out_size; (void)ws_size;
  const float* x  = (const float*)d_in[0];
  const float* WQ = (const float*)d_in[1];
  const float* WK = (const float*)d_in[2];
  const float* WV = (const float*)d_in[3];
  const float* WO = (const float*)d_in[4];

  char* ws = (char*)d_ws;
  size_t off = 0;
  const size_t XE = (size_t)MTOT * D_MODEL;     // 4M elems
  const size_t WE = (size_t)D_MODEL * D_MODEL;  // 1M elems
  __bf16* xb  = (__bf16*)(ws + off); off += XE * 2;
  __bf16* wqb = (__bf16*)(ws + off); off += WE * 2;
  __bf16* wkb = (__bf16*)(ws + off); off += WE * 2;
  __bf16* wvb = (__bf16*)(ws + off); off += WE * 2;
  __bf16* wob = (__bf16*)(ws + off); off += WE * 2;
  __bf16* Qb  = (__bf16*)(ws + off); off += XE * 2;   // [B,H,S,64]
  __bf16* Kb  = (__bf16*)(ws + off); off += XE * 2;
  __bf16* Vb  = (__bf16*)(ws + off); off += XE * 2;
  __bf16* Ob  = (__bf16*)(ws + off); off += XE * 2;   // [B*S, E]

  cvt_f32_bf16<<<(int)((XE + 255) / 256), 256, 0, stream>>>(x,  xb,  (int)XE);
  cvt_f32_bf16<<<(int)((WE + 255) / 256), 256, 0, stream>>>(WQ, wqb, (int)WE);
  cvt_f32_bf16<<<(int)((WE + 255) / 256), 256, 0, stream>>>(WK, wkb, (int)WE);
  cvt_f32_bf16<<<(int)((WE + 255) / 256), 256, 0, stream>>>(WV, wvb, (int)WE);
  cvt_f32_bf16<<<(int)((WE + 255) / 256), 256, 0, stream>>>(WO, wob, (int)WE);

  dim3 gg(MTOT / GT_M, D_MODEL / GT_N);   // 32 x 8
  gemm_bf16_128x128<0><<<gg, 256, 0, stream>>>(xb, wqb, (void*)Qb, MTOT, D_MODEL, D_MODEL);
  gemm_bf16_128x128<0><<<gg, 256, 0, stream>>>(xb, wkb, (void*)Kb, MTOT, D_MODEL, D_MODEL);
  gemm_bf16_128x128<0><<<gg, 256, 0, stream>>>(xb, wvb, (void*)Vb, MTOT, D_MODEL, D_MODEL);

  dim3 ga(SEQ / QBLK, BATCH * NUM_HEADS); // 32 x 32
  attn_fwd<<<ga, 128, 0, stream>>>(Qb, Kb, Vb, Ob);

  gemm_bf16_128x128<1><<<gg, 256, 0, stream>>>(Ob, wob, d_out, MTOT, D_MODEL, D_MODEL);
}
